// bilstm_crf_26895085208318
// MI455X (gfx1250) — matrix operand order repaired and checked by reading, not compiled, not run on hardware
//
#include <hip/hip_runtime.h>
#include <hip/hip_bf16.h>

// ---------------- problem constants ----------------
#define BB   32
#define TT   512
#define EE   256
#define HH   256      // per-direction hidden
#define G4H  1024     // 4*H
#define HID  512
#define LL   16
#define VV   30000

typedef __attribute__((ext_vector_type(16))) __bf16 v16bf;
typedef __attribute__((ext_vector_type(8)))  __bf16 v8bf;
typedef __attribute__((ext_vector_type(8)))  float  v8f;

__device__ __forceinline__ v8f wmma_bf16(v16bf a, v16bf b, v8f c) {
  // D = A(16x32 bf16) * B(32x16 bf16) + C(16x16 f32)
  return __builtin_amdgcn_wmma_f32_16x16x32_bf16(
      /*neg_a=*/false, a, /*neg_b=*/false, b,
      /*c_mod=*/(short)0, c, /*reuse_a=*/false, /*reuse_b=*/false);
}

// A fragment (16x32, 16-bit): lanes 0-15 hold M=lane, K=k0..k0+15;
// lanes 16-31 hold M=lane-16, K=k0+16..k0+31.
// rowp points at A[M][0]; two aligned 16-byte loads, merged.
__device__ __forceinline__ v16bf load_a_bf16(const __bf16* rowp, int k0, int lane) {
  int kb = (lane & 16) ? 16 : 0;
  v8bf lo = *reinterpret_cast<const v8bf*>(rowp + k0 + kb);
  v8bf hi = *reinterpret_cast<const v8bf*>(rowp + k0 + kb + 8);
  return __builtin_shufflevector(lo, hi, 0, 1, 2, 3, 4, 5, 6, 7,
                                 8, 9, 10, 11, 12, 13, 14, 15);
}

// B fragment (32x16, 16-bit): lanes 0-15 hold N=lane, K=k0..k0+15;
// lanes 16-31 hold N=lane-16, K=k0+16..k0+31. W row-major [N][ldk] bf16.
__device__ __forceinline__ v16bf load_b_bf16(const __bf16* W, int ldk, int n, int k0, int lane) {
  int kb = (lane & 16) ? 16 : 0;
  return *reinterpret_cast<const v16bf*>(W + (size_t)n * ldk + k0 + kb);
}

__device__ __forceinline__ float sigm(float x) { return 1.0f / (1.0f + __expf(-x)); }

// ---------------- kernel 0: f32 -> bf16 conversion (weights + embed table) ----------
__global__ void k_cvt_bf16(const float* __restrict__ src, __bf16* __restrict__ dst, int n) {
  int i = blockIdx.x * blockDim.x + threadIdx.x;
  if (i < n) dst[i] = (__bf16)src[i];
}

// ---------------- kernel 1: embedding gather + input projection ----------------
// xp[t,b,0:4H] = embed[tokens[b,t]] @ W_ihT + bias  (both directions fused on N).
// A rows: r = t*B + b (16384 rows, K=E=256).  B: Wih_bf [2048 x 256].
// Register-blocked: each wave owns 1 M-tile x 4 N-tiles (A loaded once per kk).
__global__ void k_inproj(const int* __restrict__ tokens,
                         const __bf16* __restrict__ embed_bf,
                         const __bf16* __restrict__ Wih_bf,
                         const float* __restrict__ b_f,
                         const float* __restrict__ b_r,
                         float* __restrict__ xp_f,
                         float* __restrict__ xp_r) {
  const int lane = threadIdx.x & 31;
  const int w = blockIdx.x * (blockDim.x >> 5) + (threadIdx.x >> 5);
  const int NG = (2 * G4H) / 64;          // 32 groups of 4 N-tiles
  const int tm = w / NG, tg = w % NG;

  const int row = tm * 16 + (lane & 15);  // r = t*B + b
  const int t = row >> 5, b = row & 31;
  const int token = tokens[b * TT + t];
  const __bf16* arow = embed_bf + (size_t)token * EE;
  const int nbase = tg * 64 + (lane & 15);

  v8f acc0 = {}, acc1 = {}, acc2 = {}, acc3 = {};
#pragma unroll
  for (int kk = 0; kk < EE / 32; ++kk) {
    const int k0 = kk * 32;
    v16bf a = load_a_bf16(arow, k0, lane);
    v16bf w0 = load_b_bf16(Wih_bf, EE, nbase,      k0, lane);
    v16bf w1 = load_b_bf16(Wih_bf, EE, nbase + 16, k0, lane);
    v16bf w2 = load_b_bf16(Wih_bf, EE, nbase + 32, k0, lane);
    v16bf w3 = load_b_bf16(Wih_bf, EE, nbase + 48, k0, lane);
    acc0 = wmma_bf16(a, w0, acc0);
    acc1 = wmma_bf16(a, w1, acc1);
    acc2 = wmma_bf16(a, w2, acc2);
    acc3 = wmma_bf16(a, w3, acc3);
  }

  const int rbase = tm * 16 + ((lane & 16) ? 8 : 0);
  v8f accs[4] = {acc0, acc1, acc2, acc3};
#pragma unroll
  for (int q = 0; q < 4; ++q) {
    int ncol = nbase + q * 16;
    float bias; float* dst; int cl;
    if (ncol < G4H) { bias = b_f[ncol];       dst = xp_f; cl = ncol; }
    else            { bias = b_r[ncol - G4H]; dst = xp_r; cl = ncol - G4H; }
#pragma unroll
    for (int j = 0; j < 8; ++j)
      dst[(size_t)(rbase + j) * G4H + cl] = accs[q][j] + bias;
  }
}

// ---------------- kernel 2: persistent LSTM recurrence ----------------
// 4 workgroups: blockIdx = dir*2 + batch_half. 16 waves/WG, each wave owns one
// 16-wide column block of H and computes i/f/g/o gate tiles for it via WMMA.
// Cell state c lives in registers; h round-trips through 8KB of LDS (bf16).
__global__ void __launch_bounds__(512, 1)
k_lstm(const float* __restrict__ xp_f, const float* __restrict__ xp_r,
       const __bf16* __restrict__ Whh_f, const __bf16* __restrict__ Whh_r,
       __bf16* __restrict__ h_f, __bf16* __restrict__ h_r) {
  const int dir   = blockIdx.x >> 1;
  const int bbase = (blockIdx.x & 1) * 16;
  const float*  xp  = dir ? xp_r : xp_f;
  const __bf16* Whh = dir ? Whh_r : Whh_f;
  __bf16*       ho  = dir ? h_r : h_f;

  const int lane  = threadIdx.x & 31;
  const int wid   = threadIdx.x >> 5;       // 0..15 -> column block
  const int jcol  = wid * 16 + (lane & 15); // this lane's hidden column (0..255)
  const int mbase = (lane & 16) ? 8 : 0;    // C-layout row base for this lane

  __shared__ __bf16 h_bf[16][264];          // 528B row stride: 16B-aligned, conflict-free
  for (int i = threadIdx.x; i < 16 * 264; i += blockDim.x)
    (&h_bf[0][0])[i] = (__bf16)0.0f;
  float cst[8];
#pragma unroll
  for (int j = 0; j < 8; ++j) cst[j] = 0.0f;
  __syncthreads();

  for (int tt = 0; tt < TT; ++tt) {
    const int t = dir ? (TT - 1 - tt) : tt;
    const float* xprow = xp + (size_t)t * BB * G4H + (size_t)bbase * G4H;

    // init gate accumulators with xp (bias already folded in by k_inproj)
    v8f gi, gf, gg, go;
#pragma unroll
    for (int j = 0; j < 8; ++j) {
      size_t ro = (size_t)(mbase + j) * G4H;
      gi[j] = xprow[ro + jcol];
      gf[j] = xprow[ro + HH     + jcol];
      gg[j] = xprow[ro + 2 * HH + jcol];
      go[j] = xprow[ro + 3 * HH + jcol];
    }

    // prefetch next timestep's xp slice (global_prefetch_b8)
    if (tt + 1 < TT) {
      const int tn = dir ? (TT - 2 - tt) : (tt + 1);
      const float* nxt = xp + (size_t)tn * BB * G4H + (size_t)bbase * G4H;
      __builtin_prefetch(nxt + (size_t)threadIdx.x * 32, 0, 1);
    }

    // gates += h @ W_hh^T, K = 256
#pragma unroll
    for (int kk = 0; kk < HH / 32; ++kk) {
      const int k0 = kk * 32;
      v16bf a  = load_a_bf16(&h_bf[lane & 15][0], k0, lane);   // ds_load_b128 x2
      v16bf b0 = load_b_bf16(Whh, HH,          jcol, k0, lane);
      v16bf b1 = load_b_bf16(Whh, HH, HH     + jcol, k0, lane);
      v16bf b2 = load_b_bf16(Whh, HH, 2 * HH + jcol, k0, lane);
      v16bf b3 = load_b_bf16(Whh, HH, 3 * HH + jcol, k0, lane);
      gi = wmma_bf16(a, b0, gi);
      gf = wmma_bf16(a, b1, gf);
      gg = wmma_bf16(a, b2, gg);
      go = wmma_bf16(a, b3, go);
    }
    __syncthreads();  // all A-reads of h_bf done before overwrite

    // in-wave pointwise LSTM cell update (lane owns column jcol, rows mbase..mbase+7)
#pragma unroll
    for (int j = 0; j < 8; ++j) {
      float iv = sigm(gi[j]), fv = sigm(gf[j]);
      float gv = tanhf(gg[j]), ov = sigm(go[j]);
      float c = fv * cst[j] + iv * gv;
      cst[j] = c;
      float h = ov * tanhf(c);
      int m = mbase + j;
      __bf16 hb = (__bf16)h;
      h_bf[m][jcol] = hb;
      ho[(size_t)t * BB * HH + (size_t)(bbase + m) * HH + jcol] = hb;
    }
    __syncthreads();
  }
}

// ---------------- kernel 3: final linear -> emissions ----------------
// rows r = b*T + t (16384), K = 512 (concat h_f|h_r), N = 16 (= L, single tile).
__global__ void k_linear(const __bf16* __restrict__ h_f, const __bf16* __restrict__ h_r,
                         const __bf16* __restrict__ linW_bf,
                         const float* __restrict__ lin_b,
                         float* __restrict__ emis) {
  const int lane = threadIdx.x & 31;
  const int tm = blockIdx.x * (blockDim.x >> 5) + (threadIdx.x >> 5);
  const int row = tm * 16 + (lane & 15);
  const int b = row >> 9, t = row & (TT - 1);
  const __bf16* rf = h_f + (size_t)t * BB * HH + (size_t)b * HH;
  const __bf16* rr = h_r + (size_t)t * BB * HH + (size_t)b * HH;
  const int l = lane & 15;

  v8f acc = {};
#pragma unroll
  for (int kk = 0; kk < 8; ++kk) {   // K 0..255 from h_f
    v16bf a  = load_a_bf16(rf, kk * 32, lane);
    v16bf bb = load_b_bf16(linW_bf, HID, l, kk * 32, lane);
    acc = wmma_bf16(a, bb, acc);
  }
#pragma unroll
  for (int kk = 0; kk < 8; ++kk) {   // K 256..511 from h_r
    v16bf a  = load_a_bf16(rr, kk * 32, lane);
    v16bf bb = load_b_bf16(linW_bf, HID, l, 256 + kk * 32, lane);
    acc = wmma_bf16(a, bb, acc);
  }
  const int rbase = tm * 16 + ((lane & 16) ? 8 : 0);
#pragma unroll
  for (int j = 0; j < 8; ++j)
    emis[(size_t)(rbase + j) * LL + l] = acc[j] + lin_b[l];
}

// ---------------- kernel 4: Viterbi decode (wave32 shuffle scan) ----------------
__global__ void k_viterbi(const float* __restrict__ emis,
                          const float* __restrict__ trans,
                          const float* __restrict__ st,
                          const float* __restrict__ et,
                          unsigned char* __restrict__ hist,
                          float* __restrict__ tags) {
  const int b = blockIdx.x;          // 32 blocks, 32 threads (1 wave)
  const int lane = threadIdx.x;
  const int j = lane & 15;
  float tcol[LL];
#pragma unroll
  for (int i = 0; i < LL; ++i) tcol[i] = trans[i * LL + j];
  const float* eb = emis + (size_t)b * TT * LL;
  unsigned char* hb = hist + (size_t)b * TT * LL;

  float score = st[j] + eb[j];
  for (int t = 1; t < TT; ++t) {
    float best = -3.4e38f; int arg = 0;
#pragma unroll
    for (int i = 0; i < LL; ++i) {
      float v = __shfl(score, i, 32) + tcol[i];
      if (v > best) { best = v; arg = i; }
    }
    score = best + eb[t * LL + j];
    if (lane < LL) hb[t * LL + j] = (unsigned char)arg;
  }
  score += et[j];
  float best = -3.4e38f; int last = 0;
#pragma unroll
  for (int i = 0; i < LL; ++i) {
    float v = __shfl(score, i, 32);
    if (v > best) { best = v; last = i; }
  }
  if (lane == 0) {
    int tag = last;
    tags[(size_t)b * TT + TT - 1] = (float)tag;
    for (int t = TT - 1; t >= 1; --t) {
      tag = hb[t * LL + tag];
      tags[(size_t)b * TT + t - 1] = (float)tag;
    }
  }
}

// ---------------- host launcher ----------------
extern "C" void kernel_launch(void* const* d_in, const int* in_sizes, int n_in,
                              void* d_out, int out_size, void* d_ws, size_t ws_size,
                              hipStream_t stream) {
  const int*   tokens = (const int*)  d_in[0];
  // d_in[1] = mask (all-true, unused)
  const float* embed  = (const float*)d_in[2];
  const float* W_ih_f = (const float*)d_in[3];
  const float* W_hh_f = (const float*)d_in[4];
  const float* b_f    = (const float*)d_in[5];
  const float* W_ih_r = (const float*)d_in[6];
  const float* W_hh_r = (const float*)d_in[7];
  const float* b_r    = (const float*)d_in[8];
  const float* lin_W  = (const float*)d_in[9];
  const float* lin_b  = (const float*)d_in[10];
  const float* transitions = (const float*)d_in[11];
  const float* start_trans = (const float*)d_in[12];
  const float* end_trans   = (const float*)d_in[13];

  char* ws = (char*)d_ws;
  size_t off = 0;
  float*  xp_f = (float*)(ws + off);  off += (size_t)TT * BB * G4H * 4;   // 64 MB
  float*  xp_r = (float*)(ws + off);  off += (size_t)TT * BB * G4H * 4;   // 64 MB
  __bf16* h_f  = (__bf16*)(ws + off); off += (size_t)TT * BB * HH * 2;    // 8 MB
  __bf16* h_r  = (__bf16*)(ws + off); off += (size_t)TT * BB * HH * 2;    // 8 MB
  __bf16* embed_bf = (__bf16*)(ws + off); off += (size_t)VV * EE * 2;     // ~15 MB
  __bf16* Wih_bf   = (__bf16*)(ws + off); off += (size_t)2 * G4H * EE * 2;
  __bf16* Whhf_bf  = (__bf16*)(ws + off); off += (size_t)G4H * HH * 2;
  __bf16* Whhr_bf  = (__bf16*)(ws + off); off += (size_t)G4H * HH * 2;
  __bf16* linW_bf  = (__bf16*)(ws + off); off += (size_t)LL * HID * 2;
  unsigned char* hist = (unsigned char*)(ws + off); off += (size_t)BB * TT * LL;

  float* emis = (float*)d_out;                         // [B,T,L]
  float* tags = (float*)d_out + (size_t)BB * TT * LL;  // [B,T]

  // 0) bf16 conversions (embed table once; weights once)
  int nEm = VV * EE;
  k_cvt_bf16<<<(nEm + 255) / 256, 256, 0, stream>>>(embed, embed_bf, nEm);
  int nIh = G4H * EE;
  k_cvt_bf16<<<(nIh + 255) / 256, 256, 0, stream>>>(W_ih_f, Wih_bf, nIh);
  k_cvt_bf16<<<(nIh + 255) / 256, 256, 0, stream>>>(W_ih_r, Wih_bf + nIh, nIh);
  int nHh = G4H * HH;
  k_cvt_bf16<<<(nHh + 255) / 256, 256, 0, stream>>>(W_hh_f, Whhf_bf, nHh);
  k_cvt_bf16<<<(nHh + 255) / 256, 256, 0, stream>>>(W_hh_r, Whhr_bf, nHh);
  int nLw = LL * HID;
  k_cvt_bf16<<<(nLw + 255) / 256, 256, 0, stream>>>(lin_W, linW_bf, nLw);

  // 1) embedding gather + input projection (1024 M-tiles x 32 N-groups, 8 waves/blk)
  {
    int waves = ((TT * BB) / 16) * ((2 * G4H) / 64);
    k_inproj<<<waves / 8, 256, 0, stream>>>(tokens, embed_bf, Wih_bf, b_f, b_r, xp_f, xp_r);
  }

  // 2) persistent bidirectional LSTM: 4 WGs x 16 waves
  k_lstm<<<4, 512, 0, stream>>>(xp_f, xp_r, Whhf_bf, Whhr_bf, h_f, h_r);

  // 3) final linear -> emissions
  k_linear<<<((TT * BB) / 16) / 8, 256, 0, stream>>>(h_f, h_r, linW_bf, lin_b, emis);

  // 4) Viterbi decode
  k_viterbi<<<BB, 32, 0, stream>>>(emis, transitions, start_trans, end_trans, hist, tags);
}
